// Conv_MoE_8091718385699
// MI455X (gfx1250) — compile-verified, matrix operand order checked by reading, not run on hardware
//
#include <hip/hip_runtime.h>
#include <hip/hip_bf16.h>
#include <cstdint>

typedef __attribute__((ext_vector_type(16))) __bf16 v16bf;
typedef __attribute__((ext_vector_type(8)))  float  v8f;

#define B_    32
#define CIN   128
#define COUT  128
#define HH    56
#define WW    56
#define HW    (HH*WW)
#define NEXP  8

// LDS tile geometry for the conv kernel
#define ROWS  6          // 4 output rows + 2 halo
#define COLS  58         // 56 + 2 halo
#define CPITCH 40        // 32 channels + pad (80 B pitch -> 16B aligned chunks)

// A-fragment LDS ring: 3 slots x 4 M-tiles x 32 rows x 40-ushort pitch (80 B)
#define APITCH   40                      // ushorts per row (64 B data + pad)
#define ASLICE   (32 * APITCH)           // 1280 ushorts per M-tile slice
#define ASLOT    (4 * ASLICE)            // 5120 ushorts per slot (10240 B)

__device__ __forceinline__ unsigned short f2bf(float f) {
  union { float f; unsigned u; } v; v.f = f;
  unsigned r = v.u + 0x7FFFu + ((v.u >> 16) & 1u);   // round-to-nearest-even
  return (unsigned short)(r >> 16);
}

// ---------------- 1) Global average pool: gap[b*CIN+c] ----------------
__global__ void gap_kernel(const float* __restrict__ x, float* __restrict__ gap) {
  __shared__ float red[128];
  const int bc = blockIdx.x;                 // b*CIN + c
  const float* p = x + (size_t)bc * HW;
  float s = 0.f;
  for (int i = threadIdx.x; i < HW; i += 128) s += p[i];
  red[threadIdx.x] = s;
  __syncthreads();
  for (int off = 64; off > 0; off >>= 1) {
    if (threadIdx.x < off) red[threadIdx.x] += red[threadIdx.x + off];
    __syncthreads();
  }
  if (threadIdx.x == 0) gap[bc] = red[0] * (1.0f / (float)HW);
}

// ---------------- 2) Gate: top-2 + softmax, write weights output ----------------
__global__ void gate_kernel(const float* __restrict__ gap,
                            const float* __restrict__ gw,
                            const float* __restrict__ gb,
                            float* __restrict__ wout,      // [B, NEXP] output tail
                            int*   __restrict__ sel_i,     // [B,2]
                            float* __restrict__ sel_w) {   // [B,2]
  const int b = threadIdx.x;
  if (b >= B_) return;
  float lg[NEXP];
  for (int e = 0; e < NEXP; ++e) {
    float s = gb[e];
    const float* g = gap + b * CIN;
    const float* w = gw + e * CIN;
    for (int c = 0; c < CIN; ++c) s += g[c] * w[c];
    lg[e] = s;
  }
  int i0 = 0;
  for (int e = 1; e < NEXP; ++e) if (lg[e] > lg[i0]) i0 = e;
  int i1 = -1;
  for (int e = 0; e < NEXP; ++e) {
    if (e == i0) continue;
    if (i1 < 0 || lg[e] > lg[i1]) i1 = e;
  }
  const float w1 = 1.0f / (1.0f + __expf(lg[i0] - lg[i1]));
  const float w0 = 1.0f - w1;
  for (int e = 0; e < NEXP; ++e) wout[b * NEXP + e] = 0.0f;
  wout[b * NEXP + i0] = w0;
  wout[b * NEXP + i1] = w1;
  sel_i[2 * b]     = i0;  sel_i[2 * b + 1] = i1;
  sel_w[2 * b]     = w0;  sel_w[2 * b + 1] = w1;
}

// ---------------- 3) Fold selected experts into one bf16 filter per batch ----------
// dst layout: wc[b][tap][cout][cin]  (tap = kh*3+kw)
__global__ void combine_kernel(const float* __restrict__ cw,
                               const int*   __restrict__ sel_i,
                               const float* __restrict__ sel_w,
                               unsigned short* __restrict__ wc) {
  const int t = blockIdx.x * 256 + threadIdx.x;
  if (t >= B_ * 9 * COUT * CIN) return;
  const int cin  = t & 127;
  const int t2   = t >> 7;
  const int cout = t2 & 127;
  const int t3   = t2 >> 7;
  const int tap  = t3 % 9;
  const int b    = t3 / 9;
  const int e0 = sel_i[2 * b], e1 = sel_i[2 * b + 1];
  const float w0 = sel_w[2 * b], w1 = sel_w[2 * b + 1];
  const size_t s0 = ((size_t)((e0 * COUT + cout) * CIN + cin)) * 9 + tap;
  const size_t s1 = ((size_t)((e1 * COUT + cout) * CIN + cin)) * 9 + tap;
  wc[t] = f2bf(w0 * cw[s0] + w1 * cw[s1]);
}

// ---------------- 4) Implicit-GEMM conv via bf16 WMMA + async A staging ----------
// Block: batch b, 4-row output strip (N = 224 px = 14 subtiles of 16).
// 8 waves = 4 M-tiles (32 cout each) x 2 N-halves (7 subtiles each).
// A fragments are staged global->LDS with GLOBAL_LOAD_ASYNC_TO_LDS_B128 one tap
// ahead into a 3-slot ring (slot = tap%3, compile-time since 9%3==0); the async
// copies carry no VGPR results so the scheduler cannot sink them. Consumption
// gated by s_wait_asynccnt (wave-private, in-order).
__global__ __launch_bounds__(256) void moe_conv_kernel(
    const float* __restrict__ x,
    const unsigned short* __restrict__ wc,
    float* __restrict__ out) {
  __shared__ __align__(16) unsigned short lds_x[ROWS * COLS * CPITCH];
  __shared__ __align__(16) unsigned short lds_a[3 * ASLOT];

  const int tid    = threadIdx.x;
  const int b      = blockIdx.x / 14;
  const int hs     = blockIdx.x % 14;
  const int h_base = hs * 4;
  const int wave   = tid >> 5;
  const int lane   = tid & 31;
  const int m0     = (wave & 3) << 5;   // 0,32,64,96
  const int nhalf  = wave >> 2;         // 0 or 1
  const int lhalf  = lane >> 4;         // 0: lanes 0-15, 1: lanes 16-31
  const int nlane  = lane & 15;
  const int kb     = lhalf << 3;        // K sub-offset {0,8}

  // Per-lane LDS element base (tap 0) for each of the 7 N-subtiles
  int ebase[7];
#pragma unroll
  for (int nt = 0; nt < 7; ++nt) {
    const int n    = nhalf * 112 + nt * 16 + nlane;  // pixel within 4-row strip
    const int hloc = n / WW;
    const int w    = n - hloc * WW;
    ebase[nt] = (hloc * COLS + w) * CPITCH + kb;
  }

  v8f zero = {};
  v8f acc[2][7];
#pragma unroll
  for (int mt = 0; mt < 2; ++mt)
#pragma unroll
    for (int i = 0; i < 7; ++i) acc[mt][i] = zero;

  const float* xb = x + (size_t)b * CIN * HW;
  const unsigned short* wgbase = wc + (size_t)b * 9 * COUT * CIN;  // SGPR base

  // Async copy destinations: LDS byte offsets, one per ring slot.
  const unsigned abase0 =
      (unsigned)(uintptr_t)(&lds_a[0]) + ((wave & 3) * ASLICE + lane * APITCH) * 2;
  // Global byte offset of this lane's A row (tap 0, c0 = 0)
  const unsigned vrow = (unsigned)((m0 + lane) * CIN) * 2u;

  // Issue 4 x b128 async copies: one 64-B weight row per lane into its LDS row.
  auto issueA = [&](unsigned lds_dst, unsigned voff) {
    asm volatile("global_load_async_to_lds_b128 %0, %1, %2 offset:0"
                 :: "v"(lds_dst), "v"(voff), "s"(wgbase) : "memory");
    asm volatile("global_load_async_to_lds_b128 %0, %1, %2 offset:16"
                 :: "v"(lds_dst), "v"(voff), "s"(wgbase) : "memory");
    asm volatile("global_load_async_to_lds_b128 %0, %1, %2 offset:32"
                 :: "v"(lds_dst), "v"(voff), "s"(wgbase) : "memory");
    asm volatile("global_load_async_to_lds_b128 %0, %1, %2 offset:48"
                 :: "v"(lds_dst), "v"(voff), "s"(wgbase) : "memory");
  };

  union AB { uint4 q[2]; v16bf v; };

  // Cold start: stage (c0=0, tap=0) into slot 0; latency hidden by first fill.
  issueA(abase0, vrow);

  for (int c0 = 0; c0 < CIN; c0 += 32) {
    const unsigned vb  = vrow + (unsigned)c0 * 2u;                    // this block
    const unsigned vbn = vrow + (unsigned)(((c0 + 32) & 127) * 2);    // next block

    __syncthreads();
    // cooperative fill: 8 groups of 4 channels x 6 rows x 58 cols; ds_store_b64
    for (int idx = tid; idx < 8 * ROWS * COLS; idx += 256) {
      const int cg  = idx / (ROWS * COLS);
      const int rem = idx % (ROWS * COLS);
      const int r   = rem / COLS;
      const int col = rem % COLS;
      const int h_in = h_base - 1 + r;
      const int w_in = col - 1;
      const int c    = cg << 2;
      unsigned lo = 0, hi = 0;
      if (h_in >= 0 && h_in < HH && (unsigned)w_in < (unsigned)WW) {
        const float* p = xb + (size_t)(c0 + c) * HW + h_in * WW + w_in;
        lo = (unsigned)f2bf(p[0]) | ((unsigned)f2bf(p[HW]) << 16);
        hi = (unsigned)f2bf(p[2 * HW]) | ((unsigned)f2bf(p[3 * HW]) << 16);
      }
      *(uint2*)&lds_x[(r * COLS + col) * CPITCH + c] = make_uint2(lo, hi);
    }
    __syncthreads();

#pragma unroll
    for (int tap = 0; tap < 9; ++tap) {
      // Prefetch next tap's A slice one tap ahead (ring slot (tap+1)%3; at tap 8
      // prefetch the next c-block's tap 0 into slot 0 — last read at tap 6).
      if (tap < 8) {
        issueA(abase0 + (((tap + 1) % 3) * ASLOT) * 2,
               vb + (unsigned)((tap + 1) * COUT * CIN * 2));
      } else {
        issueA(abase0, vbn);
      }
      // The 4 copies just issued are the only ones allowed outstanding; the
      // previous tap's copies (this tap's operands) are older -> complete.
      asm volatile("s_wait_asynccnt 0x4" ::: "memory");

      // A fragments from the LDS ring (conflict-free: 80-B row pitch)
      const unsigned short* ab =
          lds_a + ((tap % 3) * ASLOT + (wave & 3) * ASLICE + nlane * APITCH + kb);
      AB A0, A1;
      A0.q[0] = *(const uint4*)(ab);
      A0.q[1] = *(const uint4*)(ab + 16);
      A1.q[0] = *(const uint4*)(ab + 16 * APITCH);
      A1.q[1] = *(const uint4*)(ab + 16 * APITCH + 16);

      const int dh = tap / 3, dw = tap % 3;
      const int tapoff = (dh * COLS + dw) * CPITCH;

      // B double-buffer: prefetch nt+1 while multiplying nt
      AB Bf[2];
      {
        const int e = ebase[0] + tapoff;
        Bf[0].q[0] = *(const uint4*)(&lds_x[e]);
        Bf[0].q[1] = *(const uint4*)(&lds_x[e + 16]);
      }
#pragma unroll
      for (int nt = 0; nt < 7; ++nt) {
        if (nt < 6) {
          const int e = ebase[nt + 1] + tapoff;
          Bf[(nt + 1) & 1].q[0] = *(const uint4*)(&lds_x[e]);
          Bf[(nt + 1) & 1].q[1] = *(const uint4*)(&lds_x[e + 16]);
        }
        acc[0][nt] = __builtin_amdgcn_wmma_f32_16x16x32_bf16(
            false, A0.v, false, Bf[nt & 1].v, (short)0, acc[0][nt], false, false);
        acc[1][nt] = __builtin_amdgcn_wmma_f32_16x16x32_bf16(
            false, A1.v, false, Bf[nt & 1].v, (short)0, acc[1][nt], false, false);
      }
    }
  }

  // Epilogue: C/D layout -> lanes 0-15 hold M=r, lanes 16-31 hold M=8+r
  float* ob = out + (size_t)b * COUT * HW + h_base * WW;
  const int nbase = nhalf * 112;
#pragma unroll
  for (int mt = 0; mt < 2; ++mt) {
#pragma unroll
    for (int r = 0; r < 8; ++r) {
      const int cout = m0 + (mt << 4) + r + (lhalf << 3);
#pragma unroll
      for (int nt = 0; nt < 7; ++nt) {
        ob[(size_t)cout * HW + nbase + nt * 16 + nlane] = acc[mt][nt][r];
      }
    }
  }
}

// ---------------- launcher ----------------
extern "C" void kernel_launch(void* const* d_in, const int* in_sizes, int n_in,
                              void* d_out, int out_size, void* d_ws, size_t ws_size,
                              hipStream_t stream) {
  const float* x      = (const float*)d_in[0];
  const float* conv_w = (const float*)d_in[1];
  const float* gate_w = (const float*)d_in[2];
  const float* gate_b = (const float*)d_in[3];
  // d_in[4] = top_k (assumed 2, matching reference setup)

  float* out  = (float*)d_out;
  float* wout = out + (size_t)B_ * COUT * HW;       // weights tail of output tuple

  float*          gap   = (float*)d_ws;                                  // 16 KB
  int*            sel_i = (int*)((char*)d_ws + 16384);                   // 256 B
  float*          sel_w = (float*)((char*)d_ws + 16384 + 256);           // 256 B
  unsigned short* wcomb = (unsigned short*)((char*)d_ws + 32768);        // 9.4 MB

  gap_kernel<<<B_ * CIN, 128, 0, stream>>>(x, gap);
  gate_kernel<<<1, 32, 0, stream>>>(gap, gate_w, gate_b, wout, sel_i, sel_w);

  const int ncomb = B_ * 9 * COUT * CIN;
  combine_kernel<<<(ncomb + 255) / 256, 256, 0, stream>>>(conv_w, sel_i, sel_w, wcomb);

  moe_conv_kernel<<<B_ * 14, 256, 0, stream>>>(x, wcomb, out);
}
